// SpatioTemporalFusion_83760452207309
// MI455X (gfx1250) — matrix operand order repaired and checked by reading, not compiled, not run on hardware
//
#include <hip/hip_runtime.h>

#define BM 128
#define BN 128
#define BK 64
#define LSTR 72   // LDS row stride in ushorts (64 data + 8 pad; 144B, 16B aligned)

typedef __attribute__((ext_vector_type(16))) __bf16        v16bf;
typedef __attribute__((ext_vector_type(8)))  float         v8f;
typedef __attribute__((ext_vector_type(4)))  float         f32x4;
typedef __attribute__((ext_vector_type(4)))  unsigned int  u32x4;
typedef __attribute__((ext_vector_type(4)))  int           i32x4;
typedef unsigned short u16;

#define AS1 __attribute__((address_space(1)))
#define AS3 __attribute__((address_space(3)))

#if __has_builtin(__builtin_amdgcn_global_load_async_to_lds_b128)
#define HAVE_ASYNC_LDS 1
#else
#define HAVE_ASYNC_LDS 0
#endif

__device__ __forceinline__ void wait_async0() {
#if __has_builtin(__builtin_amdgcn_s_wait_asynccnt)
  __builtin_amdgcn_s_wait_asynccnt(0);
#else
  asm volatile("s_wait_asynccnt 0x0" ::: "memory");
#endif
}

__device__ __forceinline__ unsigned int pack2bf(float a, float b) {
  unsigned int ua = __builtin_bit_cast(unsigned int, a);
  unsigned int ub = __builtin_bit_cast(unsigned int, b);
  ua = (ua + 0x7FFFu + ((ua >> 16) & 1u)) >> 16;   // round-to-nearest-even bf16
  ub = (ub + 0x7FFFu + ((ub >> 16) & 1u)) >> 16;
  return (ua & 0xFFFFu) | (ub << 16);
}

__device__ __forceinline__ u16 f2bf(float a) {
  unsigned int ua = __builtin_bit_cast(unsigned int, a);
  ua = (ua + 0x7FFFu + ((ua >> 16) & 1u)) >> 16;
  return (u16)ua;
}

union FragAB { u32x4 u[2]; v16bf v; };

// fp32 -> packed bf16 conversion, 8 elements per thread. n must be a multiple of 2048.
__global__ __launch_bounds__(256) void cvt_f32_bf16(
    const float* __restrict__ src, u16* __restrict__ dst)
{
  size_t i = ((size_t)blockIdx.x * 256 + threadIdx.x) * 8;
  f32x4 a = *(const f32x4*)(src + i);
  f32x4 b = *(const f32x4*)(src + i + 4);
  u32x4 u;
  u.x = pack2bf(a.x, a.y); u.y = pack2bf(a.z, a.w);
  u.z = pack2bf(b.x, b.y); u.w = pack2bf(b.z, b.w);
  *(u32x4*)(dst + i) = u;
}

// C[M,N] = act( A[M,K] @ W[N,K]^T + bias ); A,W bf16; C bf16 or fp32.
// A may be split in K at Ksplit (A0|A1). Double-buffered LDS pipeline; the fill
// uses gfx1250 async global->LDS copies (ASYNCcnt) when the builtin exists.
template<bool RELU, bool OUT_BF16>
__global__ __launch_bounds__(256) void gemm_bf16(
    const u16* __restrict__ A0, int lda0,
    const u16* __restrict__ A1, int lda1, int Ksplit,
    const u16* __restrict__ W,  int ldw,
    const float* __restrict__ bias,
    void* __restrict__ Cout, int ldc, int K)
{
  __shared__ __align__(16) u16 sA[2][BM * LSTR];
  __shared__ __align__(16) u16 sB[2][BN * LSTR];

  const int tid    = threadIdx.x;
  const int blockM = blockIdx.y * BM;
  const int blockN = blockIdx.x * BN;

  const int lane = tid & 31;
  const int wid  = tid >> 5;
  const int wm   = wid >> 1;       // 0..3 -> rows wm*32
  const int wn   = wid & 1;        // 0..1 -> cols wn*64
  const int ml   = lane & 15;
  const int hi   = lane >> 4;      // lane half
  const int khA  = hi * 8;         // A frag K-half offset (ISA 16-bit A 16x32 layout)
  const int khB  = hi * 8;         // B frag K-half offset (same K order as A)

  const int frow = tid >> 1;            // 0..127 (fill row)
  const int fcol = (tid & 1) * 32;      // 0 / 32 (fill col, elements; 64B each half-row)

  v8f acc[2][4];
#pragma unroll
  for (int mt = 0; mt < 2; ++mt)
#pragma unroll
    for (int nt = 0; nt < 4; ++nt)
      acc[mt][nt] = (v8f){0.f,0.f,0.f,0.f,0.f,0.f,0.f,0.f};

  // per-tile global source addresses for this thread
  auto srcA = [&](int k0) -> const u16* {
    if (k0 < Ksplit) return A0 + (size_t)(blockM + frow) * lda0 + (k0 + fcol);
    return A1 + (size_t)(blockM + frow) * lda1 + (k0 - Ksplit + fcol);
  };
  auto srcW = [&](int k0) -> const u16* {
    return W + (size_t)(blockN + frow) * ldw + (k0 + fcol);
  };

  const int ntiles = K / BK;

  auto computeTile = [&](const u16* a, const u16* b) {
#pragma unroll
    for (int ks = 0; ks < BK; ks += 32) {
      FragAB af[2], bf[4];
#pragma unroll
      for (int mt = 0; mt < 2; ++mt) {
        int base = (wm*32 + mt*16 + ml) * LSTR + ks + khA;
        af[mt].u[0] = *(const u32x4*)&a[base];        // K kb..kb+7
        af[mt].u[1] = *(const u32x4*)&a[base + 16];   // K kb+16..kb+23
      }
#pragma unroll
      for (int nt = 0; nt < 4; ++nt) {
        int base = (wn*64 + nt*16 + ml) * LSTR + ks + khB;
        bf[nt].u[0] = *(const u32x4*)&b[base];        // K kb..kb+7
        bf[nt].u[1] = *(const u32x4*)&b[base + 16];   // K kb+16..kb+23
      }
#pragma unroll
      for (int mt = 0; mt < 2; ++mt)
#pragma unroll
        for (int nt = 0; nt < 4; ++nt)
          acc[mt][nt] = __builtin_amdgcn_wmma_f32_16x16x32_bf16(
              false, af[mt].v, false, bf[nt].v, (short)0, acc[mt][nt], false, false);
    }
  };

#if HAVE_ASYNC_LDS
  // ---- async global->LDS double-buffered pipeline (ASYNCcnt) ----
  auto issueTile = [&](int t, int buf) {
    const int k0 = t * BK;
    const u16* ga = srcA(k0);
    const u16* gw = srcW(k0);
    u16* la = &sA[buf][frow * LSTR + fcol];
    u16* lw = &sB[buf][frow * LSTR + fcol];
#pragma unroll
    for (int j = 0; j < 4; ++j) {
      __builtin_amdgcn_global_load_async_to_lds_b128(
          (AS1 i32x4*)(ga + 8*j), (AS3 i32x4*)(la + 8*j), 0, 0);
      __builtin_amdgcn_global_load_async_to_lds_b128(
          (AS1 i32x4*)(gw + 8*j), (AS3 i32x4*)(lw + 8*j), 0, 0);
    }
    if (k0 + BK < K) {        // pull the tile after next toward L2
      __builtin_prefetch(srcA(k0 + BK), 0, 1);
      __builtin_prefetch(srcW(k0 + BK), 0, 1);
    }
  };

  issueTile(0, 0);
  wait_async0();
  __syncthreads();

  for (int t = 0; t < ntiles; ++t) {
    const int cur  = t & 1;
    const bool more = (t + 1) < ntiles;
    if (more) issueTile(t + 1, cur ^ 1);   // copy engine fills other buffer
    computeTile(sA[cur], sB[cur]);         // WMMAs run from current buffer
    if (more) wait_async0();
    __syncthreads();
  }
#else
  // ---- register-staged double-buffered pipeline ----
  u32x4 pa[4], pw[4];
  auto loadTile = [&](int t) {
    const int k0 = t * BK;
    const u32x4* ga = (const u32x4*)srcA(k0);
    const u32x4* gw = (const u32x4*)srcW(k0);
#pragma unroll
    for (int j = 0; j < 4; ++j) { pa[j] = ga[j]; pw[j] = gw[j]; }
    if (k0 + BK < K) {
      __builtin_prefetch(srcA(k0 + BK), 0, 1);
      __builtin_prefetch(srcW(k0 + BK), 0, 1);
    }
  };
  auto storeTile = [&](int buf) {
    u32x4* da = (u32x4*)&sA[buf][frow * LSTR + fcol];
    u32x4* dw = (u32x4*)&sB[buf][frow * LSTR + fcol];
#pragma unroll
    for (int j = 0; j < 4; ++j) { da[j] = pa[j]; dw[j] = pw[j]; }
  };

  loadTile(0);
  storeTile(0);
  __syncthreads();

  for (int t = 0; t < ntiles; ++t) {
    const int cur  = t & 1;
    const bool more = (t + 1) < ntiles;
    if (more) loadTile(t + 1);
    computeTile(sA[cur], sB[cur]);
    if (more) storeTile(cur ^ 1);
    __syncthreads();
  }
#endif

  // Epilogue: bias (+ReLU), store. C/D layout: VGPR r -> row r + 8*hi, col = lane&15.
#pragma unroll
  for (int nt = 0; nt < 4; ++nt) {
    int col = blockN + wn*64 + nt*16 + ml;
    float bv = bias[col];
#pragma unroll
    for (int mt = 0; mt < 2; ++mt) {
      int rbase = blockM + wm*32 + mt*16 + hi*8;
#pragma unroll
      for (int r = 0; r < 8; ++r) {
        float v = acc[mt][nt][r] + bv;
        if (RELU) v = v > 0.f ? v : 0.f;
        if (OUT_BF16) ((u16*)Cout)[(size_t)(rbase + r) * ldc + col] = f2bf(v);
        else          ((float*)Cout)[(size_t)(rbase + r) * ldc + col] = v;
      }
    }
  }
}

// In-place LayerNorm over 1024 cols; one wave32 per row, shuffle reduction.
__global__ __launch_bounds__(256) void layernorm_inplace(
    float* __restrict__ X, const float* __restrict__ g, const float* __restrict__ b)
{
  const int row  = blockIdx.x * 8 + (threadIdx.x >> 5);
  const int lane = threadIdx.x & 31;
  float* xr = X + (size_t)row * 1024;
  float v[32];
  float s = 0.f, ss = 0.f;
#pragma unroll
  for (int i = 0; i < 32; ++i) {
    float x = xr[lane + i*32];
    v[i] = x; s += x; ss += x*x;
  }
#pragma unroll
  for (int off = 16; off > 0; off >>= 1) {
    s  += __shfl_xor(s,  off, 32);
    ss += __shfl_xor(ss, off, 32);
  }
  float mu  = s * (1.f/1024.f);
  float var = ss * (1.f/1024.f) - mu*mu;
  float inv = rsqrtf(var + 1e-5f);
#pragma unroll
  for (int i = 0; i < 32; ++i) {
    int c = lane + i*32;
    xr[c] = (v[i] - mu) * inv * g[c] + b[c];
  }
}

static inline void conv(const float* s, u16* d, size_t n, hipStream_t st) {
  cvt_f32_bf16<<<dim3((unsigned)(n / 2048)), 256, 0, st>>>(s, d);
}

static inline void run_gemm(const u16* A, const u16* W, const float* bias,
                            u16* C, int M, int N, int K, hipStream_t s)
{
  dim3 grid(N / BN, M / BM);
  gemm_bf16<false, true><<<grid, 256, 0, s>>>(A, K, A, K, K, W, K, bias, (void*)C, N, K);
}

extern "C" void kernel_launch(void* const* d_in, const int* in_sizes, int n_in,
                              void* d_out, int out_size, void* d_ws, size_t ws_size,
                              hipStream_t stream)
{
  (void)in_sizes; (void)n_in; (void)out_size; (void)ws_size;
  const int B = 16384, FD = 1024;
  const size_t MM = (size_t)FD * FD;        // 1M elements
  const size_t BF = (size_t)B * FD;         // 16M elements

  const float* spatial  = (const float*)d_in[0];
  const float* temporal = (const float*)d_in[1];
  const float* sp_w = (const float*)d_in[2];
  const float* sp_b = (const float*)d_in[3];
  const float* tp_w = (const float*)d_in[4];
  const float* tp_b = (const float*)d_in[5];
  // Q/K projections (6..9, 14..17, 22..25) are dead: softmax over length-1 keys == 1.
  const float* st_vw = (const float*)d_in[10];
  const float* st_vb = (const float*)d_in[11];
  const float* st_ow = (const float*)d_in[12];
  const float* st_ob = (const float*)d_in[13];
  const float* ts_vw = (const float*)d_in[18];
  const float* ts_vb = (const float*)d_in[19];
  const float* ts_ow = (const float*)d_in[20];
  const float* ts_ob = (const float*)d_in[21];
  const float* sa_vw = (const float*)d_in[26];
  const float* sa_vb = (const float*)d_in[27];
  const float* sa_ow = (const float*)d_in[28];
  const float* sa_ob = (const float*)d_in[29];
  const float* f1_w  = (const float*)d_in[30];
  const float* f1_b  = (const float*)d_in[31];
  const float* f2_w  = (const float*)d_in[32];
  const float* f2_b  = (const float*)d_in[33];
  const float* ln_g  = (const float*)d_in[34];
  const float* ln_b  = (const float*)d_in[35];

  // ---- workspace layout (bf16 elements) ----
  u16* wsp = (u16*)d_ws;            // 9 square weights + f1 (2M) = 11M elems
  u16* wtp  = wsp + MM;
  u16* wstv = wsp + 2*MM;
  u16* wsto = wsp + 3*MM;
  u16* wtsv = wsp + 4*MM;
  u16* wtso = wsp + 5*MM;
  u16* wsav = wsp + 6*MM;
  u16* wsao = wsp + 7*MM;
  u16* wf2  = wsp + 8*MM;
  u16* wf1  = wsp + 9*MM;           // 2M elems
  u16* SBF  = wsp + 11*MM;          // spatial bf16, 16M
  u16* TBF  = SBF + BF;             // temporal bf16, 16M
  u16* X    = TBF + BF;             // activations, 16M each
  u16* Y    = X + BF;
  u16* Z    = Y + BF;
  u16* D16  = (u16*)d_out;          // reuse d_out as bf16 scratch for s_att
  float* OUT = (float*)d_out;

  // ---- one-time per-launch bf16 conversion ----
  conv(spatial,  SBF,  BF, stream);
  conv(temporal, TBF,  BF, stream);
  conv(sp_w,  wsp,  MM, stream);
  conv(tp_w,  wtp,  MM, stream);
  conv(st_vw, wstv, MM, stream);
  conv(st_ow, wsto, MM, stream);
  conv(ts_vw, wtsv, MM, stream);
  conv(ts_ow, wtso, MM, stream);
  conv(sa_vw, wsav, MM, stream);
  conv(sa_ow, wsao, MM, stream);
  conv(f2_w,  wf2,  MM, stream);
  conv(f1_w,  wf1,  2*MM, stream);

  // ---- GEMM chain (all bf16 in / bf16 out except final) ----
  run_gemm(SBF, wsp,  sp_b,  X,   B, FD, FD, stream);   // sp
  run_gemm(TBF, wtp,  tp_b,  Y,   B, FD, FD, stream);   // tp
  run_gemm(Y,   wstv, st_vb, Z,   B, FD, FD, stream);   // tv = tp@st_vw
  run_gemm(Z,   wsto, st_ob, D16, B, FD, FD, stream);   // s_att -> d_out (bf16 scratch)
  run_gemm(X,   wtsv, ts_vb, Z,   B, FD, FD, stream);   // sv = sp@ts_vw
  run_gemm(Z,   wtso, ts_ob, X,   B, FD, FD, stream);   // t_att
  {
    // h = relu([s_att | t_att] @ f1_w.T + f1_b), K = 2048 split at 1024
    dim3 grid(FD / BN, B / BM);
    gemm_bf16<true, true><<<grid, 256, 0, stream>>>(
        D16, FD, X, FD, FD, wf1, 2*FD, f1_b, (void*)Y, FD, 2*FD);
  }
  run_gemm(Y, wf2,  f2_b,  Z, B, FD, FD, stream);       // fused
  run_gemm(Z, wsav, sa_vb, X, B, FD, FD, stream);       // fv
  {
    // fo = fv @ sa_ow.T + sa_ob -> fp32 into d_out
    dim3 grid(FD / BN, B / BM);
    gemm_bf16<false, false><<<grid, 256, 0, stream>>>(
        X, FD, X, FD, FD, wsao, FD, sa_ob, (void*)OUT, FD, FD);
  }
  layernorm_inplace<<<dim3(B / 8), 256, 0, stream>>>(OUT, ln_g, ln_b);
}
